// GatedAttention_12360915878686
// MI455X (gfx1250) — matrix operand order repaired and checked by reading, not compiled, not run on hardware
//
#include <hip/hip_runtime.h>

typedef __attribute__((ext_vector_type(16))) __bf16 v16bf;
typedef __attribute__((ext_vector_type(8)))  __bf16 v8bf;
typedef __attribute__((ext_vector_type(8)))  float  v8f;

union Frag16 { v16bf v; v8bf h[2]; };

__device__ __forceinline__ v8f zero8() {
    v8f z;
    #pragma unroll
    for (int i = 0; i < 8; ++i) z[i] = 0.0f;
    return z;
}

constexpr int cB   = 2;
constexpr int cS   = 2048;
constexpr int cHID = 2048;
constexpr int cH   = 16;
constexpr int cKV  = 4;
constexpr int cD   = 128;
constexpr float cEPS = 1e-6f;

// ---------------------------------------------------------------------------
// CDNA5 async global->LDS copy (16 bytes per lane), GV addressing mode.
// vdst = LDS byte address, vaddr = 64-bit global address. Tracked by ASYNCcnt.
// ---------------------------------------------------------------------------
__device__ __forceinline__ void async_copy16(uint32_t lds, uint64_t gaddr) {
    asm volatile("global_load_async_to_lds_b128 %0, %1, off"
                 :: "v"(lds), "v"(gaddr) : "memory");
}
__device__ __forceinline__ void wait_async3() {
    asm volatile("s_wait_asynccnt 0x3" ::: "memory");
}
__device__ __forceinline__ void wait_async0() {
    asm volatile("s_wait_asynccnt 0x0" ::: "memory");
}

// ---------------------------------------------------------------------------
// f32 -> bf16 straight conversion (8 elems / thread).
// ---------------------------------------------------------------------------
__global__ __launch_bounds__(256)
void cvt_bf16(const float* __restrict__ in, __bf16* __restrict__ out) {
    size_t i = ((size_t)blockIdx.x * 256 + threadIdx.x) * 8;
    float4 a = *(const float4*)(in + i);
    float4 b = *(const float4*)(in + i + 4);
    v8bf r;
    r[0] = (__bf16)a.x; r[1] = (__bf16)a.y; r[2] = (__bf16)a.z; r[3] = (__bf16)a.w;
    r[4] = (__bf16)b.x; r[5] = (__bf16)b.y; r[6] = (__bf16)b.z; r[7] = (__bf16)b.w;
    *(v8bf*)(out + i) = r;
}

// ---------------------------------------------------------------------------
// f32 [K][N] -> bf16 [N][K] transpose+convert, 32x32 LDS tiles, block (32,8).
// ---------------------------------------------------------------------------
__global__ __launch_bounds__(256)
void transpose_cvt(const float* __restrict__ in, __bf16* __restrict__ out,
                   int K, int N) {
    __shared__ float tile[32][33];
    const int n0 = blockIdx.x * 32, k0 = blockIdx.y * 32;
    const int tx = threadIdx.x, ty = threadIdx.y;
    #pragma unroll
    for (int j = 0; j < 32; j += 8)
        tile[ty + j][tx] = in[(size_t)(k0 + ty + j) * N + n0 + tx];
    __syncthreads();
    #pragma unroll
    for (int j = 0; j < 32; j += 8)
        out[(size_t)(n0 + ty + j) * K + k0 + tx] = (__bf16)tile[tx][ty + j];
}

// ---------------------------------------------------------------------------
// C(f32,MxN) = A(bf16,[M][K]) * BT(bf16,[N][K])  via WMMA.
// Block tile 128x64, BK=32, 256 threads = 8 waves (each 32x32).
// Double-buffered LDS, tiles staged with global_load_async_to_lds_b128.
// ---------------------------------------------------------------------------
__global__ __launch_bounds__(256)
void gemm_bf16_wmma(const __bf16* __restrict__ A, const __bf16* __restrict__ BT,
                    float* __restrict__ C, int M, int N, int K) {
    constexpr int BK = 32;
    __shared__ __bf16 As[2][128 * BK];   // [row][k]
    __shared__ __bf16 Bs[2][64 * BK];    // [col][k]

    const int t    = threadIdx.x;
    const int wave = t >> 5;
    const int lane = t & 31;
    const int ln   = lane & 15;
    const int hi   = lane >> 4;
    const int wm   = wave >> 1;          // 0..3
    const int wn   = wave & 1;           // 0..1
    const int blockM = blockIdx.y * 128;
    const int blockN = blockIdx.x * 64;

    const uint32_t asBase[2] = { (uint32_t)(uintptr_t)&As[0][0],
                                 (uint32_t)(uintptr_t)&As[1][0] };
    const uint32_t bsBase[2] = { (uint32_t)(uintptr_t)&Bs[0][0],
                                 (uint32_t)(uintptr_t)&Bs[1][0] };

    v8f acc[2][2];
    #pragma unroll
    for (int i = 0; i < 2; ++i)
        #pragma unroll
        for (int j = 0; j < 2; ++j) acc[i][j] = zero8();

    const int nIter = K / BK;

    // stage tile (buf, k0): A = 512 x 16B chunks, B = 256 x 16B chunks
    auto stage = [&](int buf, int k0) {
        #pragma unroll
        for (int j = 0; j < 2; ++j) {
            int c   = t + j * 256;
            int row = c >> 2;
            int off = (c & 3) * 16;
            uint64_t g = (uint64_t)(uintptr_t)(A + (size_t)(blockM + row) * K + k0) + off;
            async_copy16(asBase[buf] + c * 16, g);
        }
        {
            int c   = t;
            int row = c >> 2;
            int off = (c & 3) * 16;
            uint64_t g = (uint64_t)(uintptr_t)(BT + (size_t)(blockN + row) * K + k0) + off;
            async_copy16(bsBase[buf] + c * 16, g);
        }
    };

    stage(0, 0);
    for (int i = 0; i < nIter; ++i) {
        const int cur = i & 1;
        if (i + 1 < nIter) {
            stage(cur ^ 1, (i + 1) * BK);
            wait_async3();               // previous tile's 3 ops retired
        } else {
            wait_async0();
        }
        __syncthreads();

        const __bf16* as = &As[cur][0];
        const __bf16* bs = &Bs[cur][0];
        Frag16 afr[2], bfr[2];
        #pragma unroll
        for (int mt = 0; mt < 2; ++mt) {
            int row = wm * 32 + mt * 16 + ln;
            afr[mt].h[0] = *(const v8bf*)&as[row * BK + hi * 8];
            afr[mt].h[1] = *(const v8bf*)&as[row * BK + 16 + hi * 8];
        }
        #pragma unroll
        for (int nt = 0; nt < 2; ++nt) {
            int col = wn * 32 + nt * 16 + ln;
            bfr[nt].h[0] = *(const v8bf*)&bs[col * BK + hi * 8];
            bfr[nt].h[1] = *(const v8bf*)&bs[col * BK + 16 + hi * 8];
        }
        #pragma unroll
        for (int mt = 0; mt < 2; ++mt)
            #pragma unroll
            for (int nt = 0; nt < 2; ++nt)
                acc[mt][nt] = __builtin_amdgcn_wmma_f32_16x16x32_bf16(
                    false, afr[mt].v, false, bfr[nt].v, (short)0, acc[mt][nt],
                    false, false);
        __syncthreads();
    }

    #pragma unroll
    for (int mt = 0; mt < 2; ++mt)
        #pragma unroll
        for (int nt = 0; nt < 2; ++nt) {
            int col  = blockN + wn * 32 + nt * 16 + ln;
            int row0 = blockM + wm * 32 + mt * 16 + hi * 8;
            float* cp = C + (size_t)row0 * N + col;
            #pragma unroll
            for (int r = 0; r < 8; ++r) {
                *cp = acc[mt][nt][r];
                cp += N;
            }
        }
}

// ---------------------------------------------------------------------------
// RMSNorm + RoPE on Q and K; scatter K/V into the present cache.
// One block (128 threads) per D-row. Rows per (b,s): 16 q, 4 k, 4 v.
// ---------------------------------------------------------------------------
__global__ __launch_bounds__(128)
void qkv_post(float* __restrict__ qg, const float* __restrict__ kraw,
              const float* __restrict__ vraw, const float* __restrict__ rope,
              const float* __restrict__ qnw, const float* __restrict__ knw,
              float* __restrict__ present) {
    __shared__ float red[128];
    __shared__ float xrow[128];
    const int d  = threadIdx.x;
    const int r  = blockIdx.x % 24;
    const int bs = blockIdx.x / 24;
    const int b  = bs / cS;
    const int s  = bs % cS;

    if (r >= 20) {                 // V: plain copy into cache
        int kv = r - 20;
        float v = vraw[(size_t)bs * (cKV * cD) + kv * cD + d];
        present[((size_t)(b * 2 + 1) * cKV + kv) * cS * cD + (size_t)s * cD + d] = v;
        return;
    }

    float x;
    const float* w;
    if (r < 16) {                  // Q head
        x = qg[(size_t)bs * (cH * 2 * cD) + r * (2 * cD) + d];
        w = qnw;
    } else {                       // K head
        int kv = r - 16;
        x = kraw[(size_t)bs * (cKV * cD) + kv * cD + d];
        w = knw;
    }

    red[d] = x * x;
    __syncthreads();
    #pragma unroll
    for (int off = 64; off > 0; off >>= 1) {
        if (d < off) red[d] += red[d + off];
        __syncthreads();
    }
    float rms = rsqrtf(red[0] * (1.0f / cD) + cEPS);
    float xn  = (1.0f + w[d]) * x * rms;
    xrow[d] = xn;
    __syncthreads();

    float c  = rope[(size_t)s * 2 * cD + d];
    float sn = rope[(size_t)s * 2 * cD + cD + d];
    float rot = (d < cD / 2) ? -xrow[d + cD / 2] : xrow[d - cD / 2];
    float y = xn * c + rot * sn;

    if (r < 16) {
        qg[(size_t)bs * (cH * 2 * cD) + r * (2 * cD) + d] = y;
    } else {
        int kv = r - 16;
        present[((size_t)(b * 2 + 0) * cKV + kv) * cS * cD + (size_t)s * cD + d] = y;
    }
}

// ---------------------------------------------------------------------------
// Flash-style causal GQA attention, one wave per 16 query rows.
// Scores and P*V via bf16 WMMA; online softmax with wave32 shuffles.
// Writes gated output directly as bf16 for the Wo GEMM.
// ---------------------------------------------------------------------------
__global__ __launch_bounds__(32)
void attention_wmma(const float* __restrict__ qg,
                    const float* __restrict__ present,
                    __bf16* __restrict__ attnbf) {
    constexpr int QB = cS / 16;
    const int bid  = blockIdx.x;
    const int qb   = bid % QB;
    const int bh   = bid / QB;
    const int h    = bh % cH;
    const int b    = bh / cH;
    const int kv   = h / (cH / cKV);
    const int qBase = qb * 16;

    const int lane = threadIdx.x;
    const int ln   = lane & 15;
    const int hi   = lane >> 4;

    __shared__ __bf16 Qs[16 * 128];
    __shared__ __bf16 Ps[16 * 32];

    const float scale = 0.08838834764831845f;   // 1/sqrt(128)

    {   // stage scaled Q tile (16 x 128) into LDS as bf16
        const int row  = lane & 15;
        const int half = lane >> 4;
        const float* qp = qg + (size_t)(b * cS + qBase + row) * (cH * 2 * cD) + h * (2 * cD) + half * 64;
        #pragma unroll
        for (int i = 0; i < 64; i += 4) {
            float4 f = *(const float4*)(qp + i);
            Qs[row * 128 + half * 64 + i + 0] = (__bf16)(f.x * scale);
            Qs[row * 128 + half * 64 + i + 1] = (__bf16)(f.y * scale);
            Qs[row * 128 + half * 64 + i + 2] = (__bf16)(f.z * scale);
            Qs[row * 128 + half * 64 + i + 3] = (__bf16)(f.w * scale);
        }
    }
    __builtin_amdgcn_s_barrier();

    Frag16 qf[4];
    #pragma unroll
    for (int dc = 0; dc < 4; ++dc) {
        qf[dc].h[0] = *(const v8bf*)&Qs[ln * 128 + dc * 32 + hi * 16];
        qf[dc].h[1] = *(const v8bf*)&Qs[ln * 128 + dc * 32 + hi * 16 + 8];
    }

    const float* Kp = present + ((size_t)(b * 2 + 0) * cKV + kv) * cS * cD;
    const float* Vp = present + ((size_t)(b * 2 + 1) * cKV + kv) * cS * cD;

    v8f o[8];
    #pragma unroll
    for (int i = 0; i < 8; ++i) o[i] = zero8();
    float mrow[8], lrow[8];
    #pragma unroll
    for (int i = 0; i < 8; ++i) { mrow[i] = -1e30f; lrow[i] = 0.0f; }

    for (int kBase = 0; kBase < qBase + 16; kBase += 32) {
        v8f sacc[2];
        sacc[0] = zero8(); sacc[1] = zero8();

        #pragma unroll
        for (int tile = 0; tile < 2; ++tile) {
            const int krow = kBase + tile * 16 + ln;
            const float* kp = Kp + (size_t)krow * cD;
            #pragma unroll
            for (int dc = 0; dc < 4; ++dc) {
                Frag16 bf;
                const float* p = kp + dc * 32 + hi * 16;
                #pragma unroll
                for (int e = 0; e < 16; e += 4) {
                    float4 f = *(const float4*)(p + e);
                    bf.v[e + 0] = (__bf16)f.x;
                    bf.v[e + 1] = (__bf16)f.y;
                    bf.v[e + 2] = (__bf16)f.z;
                    bf.v[e + 3] = (__bf16)f.w;
                }
                sacc[tile] = __builtin_amdgcn_wmma_f32_16x16x32_bf16(
                    false, qf[dc].v, false, bf.v, (short)0, sacc[tile],
                    false, false);
            }
        }

        #pragma unroll
        for (int r = 0; r < 8; ++r) {
            int qrow = qBase + hi * 8 + r;
            if (kBase + ln      > qrow) sacc[0][r] = -1e30f;
            if (kBase + 16 + ln > qrow) sacc[1][r] = -1e30f;
        }

        #pragma unroll
        for (int r = 0; r < 8; ++r) {
            float v = fmaxf(sacc[0][r], sacc[1][r]);
            #pragma unroll
            for (int off = 1; off < 16; off <<= 1)
                v = fmaxf(v, __shfl_xor(v, off, 32));
            float mnew = fmaxf(mrow[r], v);
            float corr = __expf(mrow[r] - mnew);
            float p0 = __expf(sacc[0][r] - mnew);
            float p1 = __expf(sacc[1][r] - mnew);
            float rs = p0 + p1;
            #pragma unroll
            for (int off = 1; off < 16; off <<= 1)
                rs += __shfl_xor(rs, off, 32);
            lrow[r] = lrow[r] * corr + rs;
            mrow[r] = mnew;
            #pragma unroll
            for (int nt = 0; nt < 8; ++nt) o[nt][r] *= corr;

            const int row = hi * 8 + r;
            Ps[row * 32 + ln]      = (__bf16)p0;
            Ps[row * 32 + 16 + ln] = (__bf16)p1;
        }

        Frag16 pf;
        pf.h[0] = *(const v8bf*)&Ps[ln * 32 + hi * 16];
        pf.h[1] = *(const v8bf*)&Ps[ln * 32 + hi * 16 + 8];

        #pragma unroll
        for (int nt = 0; nt < 8; ++nt) {
            Frag16 vf;
            #pragma unroll
            for (int e = 0; e < 16; ++e) {
                vf.v[e] = (__bf16)Vp[(size_t)(kBase + hi * 16 + e) * cD + nt * 16 + ln];
            }
            o[nt] = __builtin_amdgcn_wmma_f32_16x16x32_bf16(
                false, pf.v, false, vf.v, (short)0, o[nt], false, false);
        }
    }

    #pragma unroll
    for (int r = 0; r < 8; ++r) {
        int srow = qBase + hi * 8 + r;
        float linv = 1.0f / lrow[r];
        const float* grow = qg + (size_t)(b * cS + srow) * (cH * 2 * cD) + h * (2 * cD) + cD;
        __bf16* orow = attnbf + (size_t)(b * cS + srow) * (cH * cD) + h * cD;
        #pragma unroll
        for (int nt = 0; nt < 8; ++nt) {
            int d = nt * 16 + ln;
            float g  = grow[d];
            float sg = 1.0f / (1.0f + __expf(-g));
            orow[d] = (__bf16)(o[nt][r] * linv * sg);
        }
    }
}

// ---------------------------------------------------------------------------
extern "C" void kernel_launch(void* const* d_in, const int* in_sizes, int n_in,
                              void* d_out, int out_size, void* d_ws, size_t ws_size,
                              hipStream_t stream) {
    (void)in_sizes; (void)n_in; (void)out_size; (void)ws_size;
    const float* hs   = (const float*)d_in[0];
    const float* rope = (const float*)d_in[2];
    const float* Wq   = (const float*)d_in[5];
    const float* Wk   = (const float*)d_in[6];
    const float* Wv   = (const float*)d_in[7];
    const float* Wo   = (const float*)d_in[8];
    const float* qnw  = (const float*)d_in[9];
    const float* knw  = (const float*)d_in[10];

    float* out     = (float*)d_out;
    float* present = out + (size_t)cB * cS * cHID;            // (B,2,KV,S,D)

    // workspace layout (bf16 region first, then f32 region)
    __bf16* hs_bf  = (__bf16*)d_ws;                           //  8,388,608
    __bf16* WqT    = hs_bf  + (size_t)8388608;                //  8,388,608
    __bf16* WkT    = WqT    + (size_t)8388608;                //  1,048,576
    __bf16* WvT    = WkT    + (size_t)1048576;                //  1,048,576
    __bf16* WoT    = WvT    + (size_t)1048576;                //  4,194,304
    __bf16* attnbf = WoT    + (size_t)4194304;                //  8,388,608
    float*  qg     = (float*)(attnbf + (size_t)8388608);      // 16,777,216 f32
    float*  kraw   = qg     + (size_t)16777216;               //  2,097,152 f32
    float*  vraw   = kraw   + (size_t)2097152;                //  2,097,152 f32

    const int M = cB * cS;                 // 4096
    dim3 blk(256);
    dim3 tblk(32, 8);

    // one-time bf16 conversion / weight transposes
    cvt_bf16<<<dim3(8388608 / (256 * 8)), blk, 0, stream>>>(hs, hs_bf);
    transpose_cvt<<<dim3(4096 / 32, 2048 / 32), tblk, 0, stream>>>(Wq, WqT, cHID, 4096);
    transpose_cvt<<<dim3(512  / 32, 2048 / 32), tblk, 0, stream>>>(Wk, WkT, cHID, 512);
    transpose_cvt<<<dim3(512  / 32, 2048 / 32), tblk, 0, stream>>>(Wv, WvT, cHID, 512);
    transpose_cvt<<<dim3(2048 / 32, 2048 / 32), tblk, 0, stream>>>(Wo, WoT, 2048, 2048);

    // projections (async-staged bf16 WMMA GEMMs)
    gemm_bf16_wmma<<<dim3(4096 / 64, M / 128), blk, 0, stream>>>(
        hs_bf, WqT, qg, M, 4096, cHID);
    gemm_bf16_wmma<<<dim3(512 / 64, M / 128), blk, 0, stream>>>(
        hs_bf, WkT, kraw, M, 512, cHID);
    gemm_bf16_wmma<<<dim3(512 / 64, M / 128), blk, 0, stream>>>(
        hs_bf, WvT, vraw, M, 512, cHID);

    qkv_post<<<dim3(M * 24), dim3(128), 0, stream>>>(
        qg, kraw, vraw, rope, qnw, knw, present);

    attention_wmma<<<dim3(cB * cH * (cS / 16)), dim3(32), 0, stream>>>(
        qg, present, attnbf);

    gemm_bf16_wmma<<<dim3(2048 / 64, M / 128), blk, 0, stream>>>(
        attnbf, WoT, out, M, 2048, cH * cD);
}
